// Block_82162724373042
// MI455X (gfx1250) — matrix operand order repaired and checked by reading, not compiled, not run on hardware
//
#include <hip/hip_runtime.h>
#include <hip/hip_bf16.h>

// ---------------- types ----------------
typedef __attribute__((ext_vector_type(16))) __bf16 v16bf;
typedef __attribute__((ext_vector_type(8)))  __bf16 v8bf;
typedef __attribute__((ext_vector_type(8)))  float  v8f;
typedef __attribute__((ext_vector_type(4)))  unsigned int v4u;
typedef __attribute__((ext_vector_type(8)))  int v8i;
typedef __attribute__((ext_vector_type(4)))  int v4i;

#define NTHREADS 256

#if __has_builtin(__builtin_amdgcn_tensor_load_to_lds)
#define HAVE_TDM 1
#endif

__device__ __forceinline__ v8f wmma_bf16(v16bf a, v16bf b, v8f c) {
  return __builtin_amdgcn_wmma_f32_16x16x32_bf16(false, a, false, b, (short)0, c, false, false);
}

__device__ __forceinline__ v16bf cat16(v8bf lo, v8bf hi) {
  v16bf r;
  #pragma unroll
  for (int i = 0; i < 8; ++i) { r[i] = lo[i]; r[8 + i] = hi[i]; }
  return r;
}

#ifdef HAVE_TDM
// TDM: DMA a [rows x kElems] bf16 tile (row stride = strideElems) from global
// into LDS at byte offset lds_off, packed row-major [rows][kElems].
// Descriptor per CDNA5 ISA ch.8: tensor declared as exactly the tile region.
__device__ __forceinline__ void tdm_load_tile(unsigned lds_off, const void* gptr,
                                              int kElems, int rows, int strideElems)
{
  unsigned long long ga = (unsigned long long)(size_t)gptr;
  v4u g0;
  g0[0] = 1u;                                               // count=1, user mode
  g0[1] = lds_off;                                          // lds_addr [63:32]
  g0[2] = (unsigned)(ga & 0xffffffffu);                     // global_addr [95:64]
  g0[3] = (unsigned)((ga >> 32) & 0x01ffffffu) | (2u << 30); // addr[56:32] | type=2
  v8i g1;
  g1[0] = (1 << 16);               // data_size=1 (2 bytes); mask/flags/pad = 0
  g1[1] = (kElems << 16);          // tensor_dim0[15:0] @bit48
  g1[2] = (rows << 16);            // tensor_dim0[31:16]=0 | tensor_dim1[15:0]
  g1[3] = (kElems << 16);          // tensor_dim1[31:16]=0 | tile_dim0
  g1[4] = rows;                    // tile_dim1 | tile_dim2=0
  g1[5] = strideElems;             // tensor_dim0_stride[31:0]
  g1[6] = 0;                       // stride0[47:32] | stride1[15:0]
  g1[7] = 0;                       // stride1[47:16]
  v4i z4 = {0, 0, 0, 0};
#if __clang_major__ >= 23
  v8i z8 = {0, 0, 0, 0, 0, 0, 0, 0};
  __builtin_amdgcn_tensor_load_to_lds(g0, g1, z4, z4, z8, 0);
#else
  __builtin_amdgcn_tensor_load_to_lds(g0, g1, z4, z4, 0);
#endif
}

__device__ __forceinline__ void tdm_wait() {
#if __has_builtin(__builtin_amdgcn_s_wait_tensorcnt)
  __builtin_amdgcn_s_wait_tensorcnt(0);
#else
  asm volatile("s_wait_tensorcnt 0x0" ::: "memory");
#endif
}
#endif  // HAVE_TDM

// ---------------- fp32 [K,N] -> bf16 [N,K] transpose-convert ----------------
#define TT 32
__global__ __launch_bounds__(NTHREADS) void cvtT_bf16_kernel(
    const float* __restrict__ in, __bf16* __restrict__ out, int K, int N)
{
  __shared__ __bf16 tile[TT][TT + 2];
  int kt = blockIdx.y * TT, nt = blockIdx.x * TT;
  int tx = threadIdx.x & 31, ty = threadIdx.x >> 5;   // 32 x 8
  #pragma unroll
  for (int i = 0; i < TT; i += 8)
    tile[ty + i][tx] = (__bf16)in[(size_t)(kt + ty + i) * N + nt + tx];
  __syncthreads();
  #pragma unroll
  for (int i = 0; i < TT; i += 8)
    out[(size_t)(nt + ty + i) * K + kt + tx] = tile[tx][ty + i];
}

// ---------------- fused LayerNorm -> bf16 ----------------
__global__ __launch_bounds__(NTHREADS) void ln_bf16_kernel(
    const float* __restrict__ x, const float* __restrict__ g, const float* __restrict__ bta,
    __bf16* __restrict__ out, int Cdim)
{
  int row = blockIdx.x;
  const float* xr = x + (size_t)row * Cdim;
  float s = 0.f, s2 = 0.f;
  for (int i = threadIdx.x; i < Cdim; i += NTHREADS) {
    float v = xr[i]; s += v; s2 += v * v;
  }
  #pragma unroll
  for (int off = 16; off > 0; off >>= 1) {
    s  += __shfl_xor(s,  off, 32);
    s2 += __shfl_xor(s2, off, 32);
  }
  __shared__ float sm[8], sq[8];
  int w = threadIdx.x >> 5;
  if ((threadIdx.x & 31) == 0) { sm[w] = s; sq[w] = s2; }
  __syncthreads();
  float ts = 0.f, tq = 0.f;
  #pragma unroll
  for (int i = 0; i < 8; ++i) { ts += sm[i]; tq += sq[i]; }
  float mean = ts / (float)Cdim;
  float var  = tq / (float)Cdim - mean * mean;
  float rstd = rsqrtf(var + 1e-5f);
  __bf16* orow = out + (size_t)row * Cdim;
  for (int i = threadIdx.x; i < Cdim; i += NTHREADS)
    orow[i] = (__bf16)((xr[i] - mean) * rstd * g[i] + bta[i]);
}

// ---------------- tiled bf16 WMMA GEMM ----------------
// C[M,N] = act(A[M,K](bf16) * Wt[N,K](bf16)^T + bias) (+resid)
// Tiles: 128x128 per block (8 waves), wave tile 32x64, BK=32.
// Tile staging via Tensor Data Mover (async DMA, TENSORcnt); fragment traffic b128.
#define BM 128
#define BN 128
#define BK 32

template<int ACT, bool RESID, bool WF32, bool WBF16>
__global__ __launch_bounds__(NTHREADS) void gemm_bf16_kernel(
    const __bf16* __restrict__ A, const __bf16* __restrict__ Wt,
    const float* __restrict__ bias, const float* __restrict__ resid,
    float* __restrict__ outF, __bf16* __restrict__ outB,
    int M, int N, int K)
{
  __shared__ __bf16 As[BM * BK];     // [m][k], row = 4 x 16B chunks
  __shared__ __bf16 Bs[BN * BK];     // [n][k]
  const int t = threadIdx.x, lane = t & 31, w = t >> 5;
  const int m0 = blockIdx.y * BM, n0 = blockIdx.x * BN;
  const int wm = (w >> 1) * 32, wn = (w & 1) * 64;
  const int half = lane >> 4, l16 = lane & 15;
#ifdef HAVE_TDM
  const unsigned As_off = (unsigned)(size_t)(void*)As;
  const unsigned Bs_off = (unsigned)(size_t)(void*)Bs;
#endif

  v8f c[2][4] = {};
  for (int k0 = 0; k0 < K; k0 += BK) {
#ifdef HAVE_TDM
    // async tile staging via TDM: wave0 -> A tile, wave1 -> B tile
    if (w == 0) {
      tdm_load_tile(As_off, &A[(size_t)m0 * K + k0], BK, BM, K);
      tdm_wait();
    } else if (w == 1) {
      tdm_load_tile(Bs_off, &Wt[(size_t)n0 * K + k0], BK, BN, K);
      tdm_wait();
    }
#else
    #pragma unroll
    for (int i = 0; i < 2; ++i) {   // A tile: 512 16B chunks, 2 per thread
      int ch = t + i * NTHREADS;
      int mm = ch >> 2, kc = (ch & 3) * 8;
      *(v8bf*)&As[mm * BK + kc] = *(const v8bf*)&A[(size_t)(m0 + mm) * K + k0 + kc];
    }
    #pragma unroll
    for (int i = 0; i < 2; ++i) {   // B tile (pre-transposed weights)
      int ch = t + i * NTHREADS;
      int nn = ch >> 2, kc = (ch & 3) * 8;
      *(v8bf*)&Bs[nn * BK + kc] = *(const v8bf*)&Wt[(size_t)(n0 + nn) * K + k0 + kc];
    }
#endif
    __syncthreads();

    v16bf a[2], b[4];
    #pragma unroll
    for (int mi = 0; mi < 2; ++mi) {
      const __bf16* p = &As[(wm + 16 * mi + l16) * BK];
      a[mi] = cat16(*(const v8bf*)&p[half * 8], *(const v8bf*)&p[16 + half * 8]);
    }
    #pragma unroll
    for (int ni = 0; ni < 4; ++ni) {
      const __bf16* p = &Bs[(wn + 16 * ni + l16) * BK];
      b[ni] = cat16(*(const v8bf*)&p[half * 8], *(const v8bf*)&p[16 + half * 8]);
    }
    #pragma unroll
    for (int mi = 0; mi < 2; ++mi)
      #pragma unroll
      for (int ni = 0; ni < 4; ++ni)
        c[mi][ni] = wmma_bf16(a[mi], b[ni], c[mi][ni]);
    __syncthreads();   // all reads done before next TDM overwrites tiles
  }

  // epilogue: C layout -> row = r + 8*(lane/16), col = lane%16
  #pragma unroll
  for (int mi = 0; mi < 2; ++mi)
    #pragma unroll
    for (int ni = 0; ni < 4; ++ni)
      #pragma unroll
      for (int r = 0; r < 8; ++r) {
        int row = m0 + wm + 16 * mi + r + 8 * half;
        int col = n0 + wn + 16 * ni + l16;
        size_t o = (size_t)row * N + col;
        float v = c[mi][ni][r] + bias[col];
        if (RESID) v += resid[o];
        if (ACT == 1) v = 0.5f * v * (1.0f + erff(v * 0.70710678118654752f));  // exact GELU
        if (WF32)  outF[o] = v;
        if (WBF16) outB[o] = (__bf16)v;
      }
}

// ---------------- flash attention (bf16 WMMA, f32 softmax) ----------------
// qkv: [B*T, 3C] bf16 (q|k|v). out: [B*T, C] bf16. One wave per 16-row Q tile.
#define NEG_BIG 1.0e30f
#define PLD 32   // P staging row stride (halves), 16B-chunk aligned

__global__ __launch_bounds__(NTHREADS) void attn_kernel(
    const __bf16* __restrict__ qkv, __bf16* __restrict__ out, int T, int Cdim)
{
  const int C3 = 3 * Cdim, hd = 64;
  __shared__ __bf16 Pl[8][16 * PLD];         // per-wave P staging (re-layout to A frag)
  const int lane = threadIdx.x & 31, w = threadIdx.x >> 5;
  const int half = lane >> 4, l16 = lane & 15;
  const int h = blockIdx.y, b = blockIdx.z;
  const int q0 = (blockIdx.x * 8 + w) * 16;
  const size_t rowbase = (size_t)b * T;

  // Q fragments: A 16x32 over d, two k-steps
  v16bf qa[2];
  #pragma unroll
  for (int ks = 0; ks < 2; ++ks) {
    const __bf16* qr = qkv + (rowbase + q0 + l16) * C3 + h * hd + ks * 32;
    #pragma unroll
    for (int i = 0; i < 8; ++i) qa[ks][i]     = qr[half * 16 + i];
    #pragma unroll
    for (int i = 0; i < 8; ++i) qa[ks][8 + i] = qr[half * 16 + 8 + i];
  }

  v8f acc[4] = {};
  float mrow[8], lrow[8];
  #pragma unroll
  for (int r = 0; r < 8; ++r) { mrow[r] = -NEG_BIG; lrow[r] = 0.f; }

  const int nIter = (q0 + 16 + 31) / 32;     // 32 keys per iteration
  for (int it = 0; it < nIter; ++it) {
    const int kt0 = it * 32;
    // S = Q * K^T : two 16-key n-tiles, K-dim 64 = 2 x 32
    v8f s[2] = {};
    #pragma unroll
    for (int nt = 0; nt < 2; ++nt) {
      #pragma unroll
      for (int ks = 0; ks < 2; ++ks) {
        v16bf kb;
        const __bf16* kr = qkv + (rowbase + kt0 + 16 * nt + l16) * C3
                           + Cdim + h * hd + ks * 32 + half * 16;
        #pragma unroll
        for (int i = 0; i < 16; ++i) kb[i] = kr[i];
        s[nt] = wmma_bf16(qa[ks], kb, s[nt]);
      }
    }
    // online softmax per row (row = r + 8*half lives across lanes of one 16-group)
    #pragma unroll
    for (int r = 0; r < 8; ++r) {
      int mg = q0 + r + 8 * half;
      float v0 = s[0][r] * 0.125f; if (kt0 + l16      > mg) v0 = -NEG_BIG;
      float v1 = s[1][r] * 0.125f; if (kt0 + 16 + l16 > mg) v1 = -NEG_BIG;
      float tmax = fmaxf(v0, v1);
      #pragma unroll
      for (int off = 1; off < 16; off <<= 1) tmax = fmaxf(tmax, __shfl_xor(tmax, off, 32));
      float mn = fmaxf(mrow[r], tmax);
      float al = __expf(mrow[r] - mn);
      float p0 = __expf(v0 - mn), p1 = __expf(v1 - mn);
      float rs = p0 + p1;
      #pragma unroll
      for (int off = 1; off < 16; off <<= 1) rs += __shfl_xor(rs, off, 32);
      lrow[r] = lrow[r] * al + rs;
      mrow[r] = mn;
      #pragma unroll
      for (int dj = 0; dj < 4; ++dj) acc[dj][r] *= al;
      Pl[w][(r + 8 * half) * PLD + l16]      = (__bf16)p0;
      Pl[w][(r + 8 * half) * PLD + 16 + l16] = (__bf16)p1;
    }
    // P -> A fragment (16x32); wave-private LDS, in-order, no barrier needed
    v16bf pa;
    {
      const __bf16* pr = &Pl[w][l16 * PLD + half * 16];
      pa = cat16(*(const v8bf*)&pr[0], *(const v8bf*)&pr[8]);
    }
    // acc += P * V : four 16-wide d tiles
    #pragma unroll
    for (int dj = 0; dj < 4; ++dj) {
      v16bf vb;
      #pragma unroll
      for (int i = 0; i < 16; ++i) {
        int key = kt0 + half * 16 + i;
        vb[i] = qkv[(rowbase + key) * C3 + 2 * Cdim + h * hd + dj * 16 + l16];
      }
      acc[dj] = wmma_bf16(pa, vb, acc[dj]);
    }
  }

  #pragma unroll
  for (int dj = 0; dj < 4; ++dj)
    #pragma unroll
    for (int r = 0; r < 8; ++r) {
      int row = q0 + r + 8 * half;
      out[(rowbase + row) * Cdim + h * hd + dj * 16 + l16] = (__bf16)(acc[dj][r] / lrow[r]);
    }
}

// ---------------- orchestration ----------------
extern "C" void kernel_launch(void* const* d_in, const int* in_sizes, int n_in,
                              void* d_out, int out_size, void* d_ws, size_t ws_size,
                              hipStream_t stream) {
  (void)in_sizes; (void)n_in; (void)out_size; (void)ws_size;
  const int Bb = 4, T = 2048, C = 1024, H = 16;
  const int M = Bb * T;                       // 8192

  const float* x      = (const float*)d_in[0];
  const float* ln1_g  = (const float*)d_in[1];
  const float* ln1_b  = (const float*)d_in[2];
  const float* W_attn = (const float*)d_in[3];
  const float* b_attn = (const float*)d_in[4];
  const float* W_proj = (const float*)d_in[5];
  const float* b_proj = (const float*)d_in[6];
  const float* ln2_g  = (const float*)d_in[7];
  const float* ln2_b  = (const float*)d_in[8];
  const float* W_fc   = (const float*)d_in[9];
  const float* b_fc   = (const float*)d_in[10];
  const float* W_fc2  = (const float*)d_in[11];
  const float* b_fc2  = (const float*)d_in[12];
  float* outp = (float*)d_out;

  char* p = (char*)d_ws;
  auto alloc = [&](size_t bytes) { char* r = p; p += (bytes + 255) & ~(size_t)255; return r; };
  __bf16* WattnT = (__bf16*)alloc((size_t)C * 3 * C * 2);   // [3C, C]
  __bf16* WprojT = (__bf16*)alloc((size_t)C * C * 2);       // [C, C]
  __bf16* WfcT   = (__bf16*)alloc((size_t)C * 4 * C * 2);   // [4C, C]
  __bf16* Wfc2T  = (__bf16*)alloc((size_t)4 * C * C * 2);   // [C, 4C]
  __bf16* xln1   = (__bf16*)alloc((size_t)M * C * 2);
  __bf16* qkv    = (__bf16*)alloc((size_t)M * 3 * C * 2);
  __bf16* attn_o = (__bf16*)alloc((size_t)M * C * 2);
  float*  x1     = (float*)alloc((size_t)M * C * 4);
  __bf16* xln2   = (__bf16*)alloc((size_t)M * C * 2);
  __bf16* hbuf   = (__bf16*)alloc((size_t)M * 4 * C * 2);

  auto cvtT = [&](const float* src, __bf16* dst, int K, int N) {  // [K,N] -> [N,K] bf16
    cvtT_bf16_kernel<<<dim3(N / TT, K / TT), NTHREADS, 0, stream>>>(src, dst, K, N);
  };
  cvtT(W_attn, WattnT, C, 3 * C);
  cvtT(W_proj, WprojT, C, C);
  cvtT(W_fc,   WfcT,   C, 4 * C);
  cvtT(W_fc2,  Wfc2T,  4 * C, C);

  // LN1
  ln_bf16_kernel<<<M, NTHREADS, 0, stream>>>(x, ln1_g, ln1_b, xln1, C);
  // qkv = xln1 @ W_attn + b_attn   -> bf16
  gemm_bf16_kernel<0, false, false, true><<<dim3(3 * C / BN, M / BM), NTHREADS, 0, stream>>>(
      xln1, WattnT, b_attn, nullptr, nullptr, qkv, M, 3 * C, C);
  // attention
  attn_kernel<<<dim3(T / 16 / 8, H, Bb), NTHREADS, 0, stream>>>(qkv, attn_o, T, C);
  // x1 = x + attn_o @ W_proj + b_proj   -> f32
  gemm_bf16_kernel<0, true, true, false><<<dim3(C / BN, M / BM), NTHREADS, 0, stream>>>(
      attn_o, WprojT, b_proj, x, x1, nullptr, M, C, C);
  // LN2
  ln_bf16_kernel<<<M, NTHREADS, 0, stream>>>(x1, ln2_g, ln2_b, xln2, C);
  // h = gelu(xln2 @ W_fc + b_fc)  -> bf16
  gemm_bf16_kernel<1, false, false, true><<<dim3(4 * C / BN, M / BM), NTHREADS, 0, stream>>>(
      xln2, WfcT, b_fc, nullptr, nullptr, hbuf, M, 4 * C, C);
  // out = x1 + h @ W_fc2 + b_fc2  -> f32
  gemm_bf16_kernel<0, true, true, false><<<dim3(C / BN, M / BM), NTHREADS, 0, stream>>>(
      hbuf, Wfc2T, b_fc2, x1, outp, nullptr, M, C, 4 * C);
}
